// RowSelfAttention_61924838474214
// MI455X (gfx1250) — matrix operand order repaired and checked by reading, not compiled, not run on hardware
//
#include <hip/hip_runtime.h>
#include <hip/hip_bf16.h>

// ---------------------------------------------------------------------------
// RowSelfAttention for MI455X (gfx1250).
// All GEMM stages -> v_wmma_f32_16x16x32_bf16 (f32 accum).
// LDS operand tiles arranged so every WMMA operand fetch is 2x ds_load_b128:
//   A tiles row-major [m][k]  : lane reads K = half*8..+7 and 16+half*8..+7
//   B tiles stored    [n][k]  : lane reads K = half*8..+7 and 16+half*8..+7
// Tile staging: GLOBAL_LOAD_ASYNC_TO_LDS_B128, double-buffered so the DMA of
// stage k+1 overlaps the WMMA of stage k (in-order ASYNCcnt retirement).
// ---------------------------------------------------------------------------

typedef __bf16 bf16_t;
typedef __attribute__((ext_vector_type(16))) __bf16 v16bf;
typedef __attribute__((ext_vector_type(8)))  __bf16 v8bf;
typedef __attribute__((ext_vector_type(4)))  __bf16 v4bf;
typedef __attribute__((ext_vector_type(8)))  float  v8f;
typedef __attribute__((ext_vector_type(4)))  int    v4i;

static constexpr int R_  = 16;
static constexpr int C_  = 256;
static constexpr int B_  = 12;
static constexpr int E_  = 768;
static constexpr int H_  = 12;
static constexpr int D_  = 64;
static constexpr int KD  = R_ * D_;        // 1024 : flattened (r,d) contraction
static constexpr int T_  = R_ * C_ * B_;   // 49152 tokens
static constexpr int NBUCK = 32;
static constexpr int AP  = 56;             // LDS tile pitch: 112B rows, 16B-aligned,
                                           // 28-dword lane stride (conflict-free)
// log(MAX_DISTANCE / max_exact) = log(50000/16) = log(3125)
#define LOG_RATIO 8.047189562f

// ---- CDNA5 async global->LDS copy (16B per lane), with sync fallback -------
#if defined(__AMDGCN__) && __has_builtin(__builtin_amdgcn_global_load_async_to_lds_b128)
#define HAVE_ASYNC_LDS 1
typedef __attribute__((address_space(1))) v4i as1_v4i;
typedef __attribute__((address_space(3))) v4i as3_v4i;
#define ASYNC_CP16(dst, src)                                                   \
  __builtin_amdgcn_global_load_async_to_lds_b128(                              \
      (as1_v4i*)(src), (as3_v4i*)(dst), 0, 0)
#if __has_builtin(__builtin_amdgcn_s_wait_asynccnt)
#define ASYNC_WAIT_N(n) __builtin_amdgcn_s_wait_asynccnt(n)
#else
#define ASYNC_WAIT_N(n) asm volatile("s_wait_asynccnt %0" ::"i"(n) : "memory")
#endif
#else
#define HAVE_ASYNC_LDS 0
#define ASYNC_CP16(dst, src) (*(uint4*)(dst) = *(const uint4*)(src))
#define ASYNC_WAIT_N(n) ((void)0)
#endif

__device__ __forceinline__ int lane_id() { return threadIdx.x & 31; }

__device__ __forceinline__ v8f wmma_bf16(v16bf a, v16bf b, v8f c) {
  return __builtin_amdgcn_wmma_f32_16x16x32_bf16(false, a, false, b, (short)0, c,
                                                 false, false);
}

// A-matrix 16x32 bf16 from LDS row-major [m][k], pitch P (elements).
template <int P>
__device__ __forceinline__ v16bf load_A(const bf16_t* s, int mbase) {
  int l = lane_id();
  int row = mbase + (l & 15);
  int half = l >> 4;
  v8bf lo = *(const v8bf*)(s + row * P + half * 8);
  v8bf hi = *(const v8bf*)(s + row * P + 16 + half * 8);
  v16bf a;
#pragma unroll
  for (int g = 0; g < 8; ++g) { a[g] = lo[g]; a[8 + g] = hi[g]; }
  return a;
}

// B-matrix 32x16 bf16 from LDS stored [n][k], pitch P.
template <int P>
__device__ __forceinline__ v16bf load_B(const bf16_t* s, int nbase) {
  int l = lane_id();
  int col = nbase + (l & 15);
  int half = l >> 4;
  v8bf lo = *(const v8bf*)(s + col * P + half * 8);
  v8bf hi = *(const v8bf*)(s + col * P + 16 + half * 8);
  v16bf b;
#pragma unroll
  for (int g = 0; g < 8; ++g) { b[g] = lo[g]; b[8 + g] = hi[g]; }
  return b;
}

// ---------------------------------------------------------------------------
// Kernel 1: W (E,E) f32 -> bf16, same row-major layout (== B-tile [n][k]).
// ---------------------------------------------------------------------------
__global__ __launch_bounds__(256) void convw_kernel(
    const float* __restrict__ wq, const float* __restrict__ wk,
    const float* __restrict__ wv, const float* __restrict__ wo,
    bf16_t* __restrict__ tq, bf16_t* __restrict__ tk,
    bf16_t* __restrict__ tv, bf16_t* __restrict__ to_) {
  int total = 4 * E_ * E_;
  for (int idx = blockIdx.x * blockDim.x + threadIdx.x; idx < total;
       idx += gridDim.x * blockDim.x) {
    int m = idx / (E_ * E_);
    int r = idx % (E_ * E_);
    const float* src;
    bf16_t* dst;
    if (m == 0)      { src = wq; dst = tq; }
    else if (m == 1) { src = wk; dst = tk; }
    else if (m == 2) { src = wv; dst = tv; }
    else             { src = wo; dst = to_; }
    dst[r] = (bf16_t)src[r];
  }
}

// ---------------------------------------------------------------------------
// Kernel 2: QKV projection, double-buffered. grid (T/32, E/64, 3), block 256.
//  q,k -> [n][h][i][r*64+d] ; v -> [n][h][r*64+d][j] (transposed).
// ---------------------------------------------------------------------------
__global__ __launch_bounds__(256) void qkv_kernel(
    const float* __restrict__ x,
    const bf16_t* __restrict__ wt_q, const bf16_t* __restrict__ wt_k,
    const bf16_t* __restrict__ wt_v,
    const float* __restrict__ bq, const float* __restrict__ bk,
    const float* __restrict__ bv,
    bf16_t* __restrict__ qb, bf16_t* __restrict__ kb, bf16_t* __restrict__ vt) {
  __shared__ __align__(16) bf16_t Xs[2][32][AP];
  __shared__ __align__(16) bf16_t Wn[2][64][AP];   // [n][k]
  int z = blockIdx.z;
  const bf16_t* wt = (z == 0) ? wt_q : (z == 1) ? wt_k : wt_v;
  const float* bias = (z == 0) ? bq : (z == 1) ? bk : bv;
  bf16_t* outp = (z == 0) ? qb : (z == 1) ? kb : vt;
  int t0 = blockIdx.x * 32;
  int n0 = blockIdx.y * 64;
  int tid = threadIdx.x;
  int w = tid >> 5, l = tid & 31;
  int mw = w >> 2, nw = w & 3;
  int xr = tid >> 3, xc = (tid & 7) * 4;   // X: 32 rows x 32 cols, 4 f32/thread
  int wr = tid >> 2, wc = (tid & 3) * 8;   // W: 64 rows x 32 cols, 8 bf16/thread

  auto stage = [&](int kb2, int buf) {
    int k0 = kb2 * 32;
    ASYNC_CP16(&Wn[buf][wr][wc], wt + (size_t)(n0 + wr) * E_ + k0 + wc);
    float4 xv = *(const float4*)(x + (size_t)(t0 + xr) * E_ + k0 + xc);
    v4bf pk;
    pk[0] = (bf16_t)xv.x; pk[1] = (bf16_t)xv.y;
    pk[2] = (bf16_t)xv.z; pk[3] = (bf16_t)xv.w;
    *(v4bf*)(&Xs[buf][xr][xc]) = pk;
  };

  v8f acc = {};
  stage(0, 0);
  for (int kb_ = 0; kb_ < E_ / 32; ++kb_) {
    int cur = kb_ & 1;
    if (kb_ + 1 < E_ / 32) { stage(kb_ + 1, 1 - cur); ASYNC_WAIT_N(1); }
    else                   { ASYNC_WAIT_N(0); }
    __syncthreads();
    v16bf a = load_A<AP>(&Xs[cur][0][0], 16 * mw);
    v16bf b = load_B<AP>(&Wn[cur][0][0], 16 * nw);
    acc = wmma_bf16(a, b, acc);
    __syncthreads();
  }
  int half = l >> 4, coln = l & 15;
  int ncol = n0 + 16 * nw + coln;
  float bval = bias[ncol];
  float scale = (z == 0) ? 0.03125f : 1.0f;   // D^-0.5 / sqrt(R) = (1/8)/4
  int h = ncol >> 6, d = ncol & 63;
  int tok0 = t0 + 16 * mw + half * 8;
  int r = tok0 / (C_ * B_);
  int rem = tok0 % (C_ * B_);
  int i = rem / B_;
  int nn = rem % B_;
#pragma unroll
  for (int p = 0; p < 8; ++p) {
    float v = (acc[p] + bval) * scale;
    size_t idx;
    if (z == 2) idx = ((size_t)(nn * H_ + h) * KD + r * D_ + d) * C_ + i;
    else        idx = ((size_t)(nn * H_ + h) * C_ + i) * KD + r * D_ + d;
    outp[idx] = (bf16_t)v;
    if (++nn == B_) { nn = 0; if (++i == C_) { i = 0; ++r; } }
  }
}

// ---------------------------------------------------------------------------
// Kernel 3: scores + T5 bucket bias + softmax, double-buffered.
// grid (H*B, C/16), block 256. Wave w owns j-tiles {16w, 16w+128}.
// attn index (h,n,i,j); bias = rel_bias[bucket(dist[h,i,j]), n]  (B==H trick).
// ---------------------------------------------------------------------------
__global__ __launch_bounds__(256) void scores_kernel(
    const bf16_t* __restrict__ qb, const bf16_t* __restrict__ kbuf,
    const int* __restrict__ distances, const float* __restrict__ rel_bias,
    float* __restrict__ probs, bf16_t* __restrict__ pb) {
  __shared__ __align__(16) bf16_t Qs[2][16][AP];
  __shared__ __align__(16) bf16_t Ks[2][256][AP];  // [j][k] == B-tile [n][k]
  __shared__ float Ss[16][256];
  __shared__ float red[16][16];
  __shared__ float rowv[16];
  __shared__ float rbias[NBUCK * H_];
  int bx = blockIdx.x;
  int h = bx / B_, n = bx % B_;
  int i0 = blockIdx.y * 16;
  int tid = threadIdx.x, w = tid >> 5, l = tid & 31;
  for (int idx = tid; idx < NBUCK * H_; idx += 256) rbias[idx] = rel_bias[idx];
  const bf16_t* Qg = qb + (size_t)((n * H_ + h) * C_ + i0) * KD;
  const bf16_t* Kg = kbuf + (size_t)(n * H_ + h) * C_ * KD;
  int qrow = tid >> 2, qcol = (tid & 3) * 8;

  auto stage = [&](int kb2, int buf) {
    int k0 = kb2 * 32;
    if (tid < 64)
      ASYNC_CP16(&Qs[buf][qrow][qcol], Qg + (size_t)qrow * KD + k0 + qcol);
#pragma unroll
    for (int it = 0; it < 4; ++it) {              // K tile 256x32
      int v = tid + it * 256;
      int rr = v >> 2, cc = (v & 3) * 8;
      ASYNC_CP16(&Ks[buf][rr][cc], Kg + (size_t)rr * KD + k0 + cc);
    }
  };

  v8f acc[2];
  acc[0] = (v8f){};
  acc[1] = (v8f){};
  stage(0, 0);
  for (int kb_ = 0; kb_ < KD / 32; ++kb_) {
    int cur = kb_ & 1;
    if (kb_ + 1 < KD / 32) { stage(kb_ + 1, 1 - cur); ASYNC_WAIT_N(4); }
    else                   { ASYNC_WAIT_N(0); }
    __syncthreads();
    v16bf a = load_A<AP>(&Qs[cur][0][0], 0);
    v16bf b0 = load_B<AP>(&Ks[cur][0][0], 16 * w);
    v16bf b1 = load_B<AP>(&Ks[cur][128][0], 16 * w);
    acc[0] = wmma_bf16(a, b0, acc[0]);
    acc[1] = wmma_bf16(a, b1, acc[1]);
    __syncthreads();
  }
  // bias + stash to LDS score board
  int half = l >> 4, coln = l & 15;
#pragma unroll
  for (int t = 0; t < 2; ++t) {
    int j = 16 * w + t * 128 + coln;
#pragma unroll
    for (int p = 0; p < 8; ++p) {
      int irow = half * 8 + p;
      int dist = distances[h * C_ * C_ + (i0 + irow) * C_ + j];
      int nn = dist < 0 ? -dist : dist;
      int bkt;
      if (nn < 16) {
        bkt = nn;
      } else {
        float nf = fmaxf((float)nn, 1.0f);
        int vl = 16 + (int)(__logf(nf * (1.0f / 16.0f)) * (15.0f / LOG_RATIO));
        bkt = vl < 31 ? vl : 31;
      }
      Ss[irow][j] = acc[t][p] + rbias[bkt * H_ + n];
    }
  }
  __syncthreads();
  // softmax over j: 16 threads per row
  int row = tid >> 4, sub = tid & 15;
  float m = -3.4e38f;
  for (int c = sub; c < 256; c += 16) m = fmaxf(m, Ss[row][c]);
  red[row][sub] = m;
  __syncthreads();
  if (sub == 0) {
    float mm = red[row][0];
    for (int s2 = 1; s2 < 16; ++s2) mm = fmaxf(mm, red[row][s2]);
    rowv[row] = mm;
  }
  __syncthreads();
  float rmax = rowv[row];
  float s = 0.f;
  for (int c = sub; c < 256; c += 16) {
    float e = __expf(Ss[row][c] - rmax);
    Ss[row][c] = e;
    s += e;
  }
  red[row][sub] = s;
  __syncthreads();
  if (sub == 0) {
    float ss = 0.f;
    for (int s2 = 0; s2 < 16; ++s2) ss += red[row][s2];
    rowv[row] = 1.0f / ss;
  }
  __syncthreads();
  float inv = rowv[row];
  size_t base = (size_t)((h * B_ + n) * C_ + i0 + row) * C_;
  for (int c = sub; c < 256; c += 16) {
    float p = Ss[row][c] * inv;
    probs[base + c] = p;          // output #2, f32 (h,n,i,j)
    pb[base + c] = (bf16_t)p;     // bf16 copy for ctx GEMM
  }
}

// ---------------------------------------------------------------------------
// Kernel 4: ctx = P (256x256) @ Vflat (256x1024) per (n,h), double-buffered.
// V stored transposed [n][h][rd][j] so the B-tile is [n][k] in LDS.
// grid (H*B, C/16, KD/128), block 256: 8 waves cover 128 output cols.
// ---------------------------------------------------------------------------
__global__ __launch_bounds__(256) void ctx_kernel(
    const bf16_t* __restrict__ pb, const bf16_t* __restrict__ vt,
    bf16_t* __restrict__ cb) {
  __shared__ __align__(16) bf16_t Ps[2][16][AP];
  __shared__ __align__(16) bf16_t Vn[2][128][AP];  // [rd][j] == [n][k]
  int bx = blockIdx.x;
  int h = bx / B_, n = bx % B_;
  int i0 = blockIdx.y * 16;
  int c0 = blockIdx.z * 128;
  int tid = threadIdx.x, w = tid >> 5, l = tid & 31;
  const bf16_t* Pg = pb + (size_t)((h * B_ + n) * C_ + i0) * C_;
  const bf16_t* Vg = vt + (size_t)(n * H_ + h) * KD * C_;
  int prow = tid >> 2, pcol = (tid & 3) * 8;

  auto stage = [&](int jb2, int buf) {
    int j0 = jb2 * 32;
    if (tid < 64)
      ASYNC_CP16(&Ps[buf][prow][pcol], Pg + (size_t)prow * C_ + j0 + pcol);
#pragma unroll
    for (int it = 0; it < 2; ++it) {              // V tile 128(rd) x 32(j)
      int v = tid + it * 256;
      int rr = v >> 2, cc = (v & 3) * 8;
      ASYNC_CP16(&Vn[buf][rr][cc], Vg + (size_t)(c0 + rr) * C_ + j0 + cc);
    }
  };

  v8f acc = {};
  stage(0, 0);
  for (int jb = 0; jb < C_ / 32; ++jb) {
    int cur = jb & 1;
    if (jb + 1 < C_ / 32) { stage(jb + 1, 1 - cur); ASYNC_WAIT_N(2); }
    else                  { ASYNC_WAIT_N(0); }
    __syncthreads();
    v16bf a = load_A<AP>(&Ps[cur][0][0], 0);
    v16bf b = load_B<AP>(&Vn[cur][0][0], 16 * w);
    acc = wmma_bf16(a, b, acc);
    __syncthreads();
  }
  int half = l >> 4, coln = l & 15;
  int col = c0 + 16 * w + coln;
#pragma unroll
  for (int p = 0; p < 8; ++p) {
    int i = i0 + half * 8 + p;
    cb[(size_t)((n * H_ + h) * C_ + i) * KD + col] = (bf16_t)acc[p];
  }
}

// ---------------------------------------------------------------------------
// Kernel 5: out = ctx @ Wo^T + bo, double-buffered. grid (T/32, E/64).
// ctx gather: within a 32-wide K chunk, d is contiguous -> 8B vector copies.
// ---------------------------------------------------------------------------
__global__ __launch_bounds__(256) void outproj_kernel(
    const bf16_t* __restrict__ cb, const bf16_t* __restrict__ wt_o,
    const float* __restrict__ bo, float* __restrict__ out) {
  __shared__ __align__(16) bf16_t As[2][32][AP];
  __shared__ __align__(16) bf16_t Wn[2][64][AP];
  int t0 = blockIdx.x * 32;
  int n0 = blockIdx.y * 64;
  int tid = threadIdx.x, w = tid >> 5, l = tid & 31;
  int mw = w >> 2, nw = w & 3;
  int wr = tid >> 2, wc = (tid & 3) * 8;
  int ar = tid >> 3, ac = (tid & 7) * 4;   // A: 32 rows x 32 cols, 4 bf16/thread
  // one token decomposition per thread (row ar is fixed)
  int tok = t0 + ar;
  int rT = tok / (C_ * B_);
  int rem = tok % (C_ * B_);
  int iT = rem / B_;
  int nT = rem % B_;

  auto stage = [&](int kb2, int buf) {
    int k0 = kb2 * 32;
    ASYNC_CP16(&Wn[buf][wr][wc], wt_o + (size_t)(n0 + wr) * E_ + k0 + wc);
    int ee = k0 + ac;
    int hh = ee >> 6, dd = ee & 63;
    const bf16_t* src =
        cb + (size_t)((nT * H_ + hh) * C_ + iT) * KD + rT * D_ + dd;
    *(uint2*)(&As[buf][ar][ac]) = *(const uint2*)src;
  };

  v8f acc = {};
  stage(0, 0);
  for (int kb_ = 0; kb_ < E_ / 32; ++kb_) {
    int cur = kb_ & 1;
    if (kb_ + 1 < E_ / 32) { stage(kb_ + 1, 1 - cur); ASYNC_WAIT_N(1); }
    else                   { ASYNC_WAIT_N(0); }
    __syncthreads();
    v16bf a = load_A<AP>(&As[cur][0][0], 16 * mw);
    v16bf b = load_B<AP>(&Wn[cur][0][0], 16 * nw);
    acc = wmma_bf16(a, b, acc);
    __syncthreads();
  }
  int half = l >> 4, coln = l & 15;
  int ncol = n0 + 16 * nw + coln;
  float bval = bo[ncol];
#pragma unroll
  for (int p = 0; p < 8; ++p) {
    int tk = t0 + 16 * mw + half * 8 + p;
    out[(size_t)tk * E_ + ncol] = acc[p] + bval;
  }
}

// ---------------------------------------------------------------------------
extern "C" void kernel_launch(void* const* d_in, const int* in_sizes, int n_in,
                              void* d_out, int out_size, void* d_ws,
                              size_t ws_size, hipStream_t stream) {
  (void)in_sizes; (void)n_in; (void)out_size; (void)ws_size;
  const float* x         = (const float*)d_in[0];
  const int*   distances = (const int*)d_in[1];
  const float* Wq = (const float*)d_in[2];
  const float* bq = (const float*)d_in[3];
  const float* Wk = (const float*)d_in[4];
  const float* bk = (const float*)d_in[5];
  const float* Wv = (const float*)d_in[6];
  const float* bv = (const float*)d_in[7];
  const float* Wo = (const float*)d_in[8];
  const float* bo = (const float*)d_in[9];
  const float* rel_bias = (const float*)d_in[10];

  float* out   = (float*)d_out;
  float* probs = out + (size_t)T_ * E_;   // second output, (H,B,C,C) f32

  char* ws = (char*)d_ws;
  size_t off = 0;
  auto alloc = [&](size_t bytes) -> char* {
    char* p = ws + off;
    off += (bytes + 255) & ~(size_t)255;
    return p;
  };
  bf16_t* tq = (bf16_t*)alloc((size_t)E_ * E_ * 2);
  bf16_t* tk = (bf16_t*)alloc((size_t)E_ * E_ * 2);
  bf16_t* tv = (bf16_t*)alloc((size_t)E_ * E_ * 2);
  bf16_t* to_ = (bf16_t*)alloc((size_t)E_ * E_ * 2);
  bf16_t* qb = (bf16_t*)alloc((size_t)B_ * H_ * C_ * KD * 2);
  bf16_t* kb = (bf16_t*)alloc((size_t)B_ * H_ * C_ * KD * 2);
  bf16_t* vtb = (bf16_t*)alloc((size_t)B_ * H_ * KD * C_ * 2);
  bf16_t* pbuf = (bf16_t*)alloc((size_t)H_ * B_ * C_ * C_ * 2);
  bf16_t* cb = (bf16_t*)alloc((size_t)B_ * H_ * C_ * KD * 2);

  convw_kernel<<<dim3(2304), dim3(256), 0, stream>>>(Wq, Wk, Wv, Wo, tq, tk, tv, to_);
  qkv_kernel<<<dim3(T_ / 32, E_ / 64, 3), dim3(256), 0, stream>>>(
      x, tq, tk, tv, bq, bk, bv, qb, kb, vtb);
  scores_kernel<<<dim3(H_ * B_, C_ / 16), dim3(256), 0, stream>>>(
      qb, kb, distances, rel_bias, probs, pbuf);
  ctx_kernel<<<dim3(H_ * B_, C_ / 16, KD / 128), dim3(256), 0, stream>>>(
      pbuf, vtb, cb);
  outproj_kernel<<<dim3(T_ / 32, E_ / 64), dim3(256), 0, stream>>>(
      cb, to_, bo, out);
}
